// BeliefPropagation_35424890257468
// MI455X (gfx1250) — compile-verified, matrix operand order checked by reading, not run on hardware
//
#include <hip/hip_runtime.h>
#include <hip/hip_bf16.h>
#include <math.h>

typedef int v8i __attribute__((ext_vector_type(8)));

// ---------------------------------------------------------------------------
// Kernel 1: column counts n1col[v] = sum_e h[e,v] via V_WMMA_I32_16X16X64_IU8.
// A = all-ones 16x64 i8 matrix, B = 64x16 tile of h (i32 {0,1} packed to i8).
// Every row of D then holds the column sums of the 64-row chunk; accumulate
// across E/64 chunks through the C operand. One wave (32 lanes) per 16 columns.
// ---------------------------------------------------------------------------
__global__ __launch_bounds__(32)
void bp_count_cols_wmma(const int* __restrict__ h, int* __restrict__ n1col,
                        int E, int V) {
    const int lane  = threadIdx.x;          // 0..31, one full wave (EXEC all 1s)
    const int vbase = blockIdx.x * 16;
    const int col   = vbase + (lane & 15);  // B-matrix N index = lane % 16
    const int half  = lane >> 4;            // lanes 16-31 carry K += 16

    v8i a;
#pragma unroll
    for (int i = 0; i < 8; ++i) a[i] = 0x01010101;  // ones, any layout works

    v8i acc = {};                                    // C starts at zero
    const int nchunk = E >> 6;                       // E/64 accumulation steps
    for (int c = 0; c < nchunk; ++c) {
        v8i bm;
#pragma unroll
        for (int q = 0; q < 8; ++q) {
            // K = (q>=4 ? 32:0) + half*16 + (q&3)*4 + byte  (any bijection of K
            // is fine for a column sum; this matches the ISA B layout).
            int ebase = c * 64 + ((q >> 2) << 5) + (half << 4) + ((q & 3) << 2);
            int b0 = h[(ebase + 0) * V + col];
            int b1 = h[(ebase + 1) * V + col];
            int b2 = h[(ebase + 2) * V + col];
            int b3 = h[(ebase + 3) * V + col];
            bm[q] = (b0 & 0xFF) | ((b1 & 0xFF) << 8) |
                    ((b2 & 0xFF) << 16) | ((b3 & 0xFF) << 24);
        }
        // 7-arg IU8 form: (sgn_a, A, sgn_b, B, C, reuse_a, reuse_b)
        acc = __builtin_amdgcn_wmma_i32_16x16x64_iu8(false, a, false, bm,
                                                     acc, false, false);
    }
    // D VGPR0, lanes 0-15 = row M=0, N=lane -> the column sum for col vbase+lane
    if (lane < 16) n1col[col] = acc[0];
}

// Integer-exponent power (negative bases legal, parity-correct; ~9 mults max).
__device__ inline float ipow(float t, int n) {
    float r = 1.0f, b = t;
    while (n > 0) { if (n & 1) r *= b; b *= b; n >>= 1; }
    return r;
}

// ---------------------------------------------------------------------------
// Kernel 2: fused 12-scalar recurrence + marginalization. One block of V
// threads. Row counts for e=0,1 computed cooperatively via LDS atomics; the
// scalar 5-iteration loop is redundantly evaluated per thread (trivial cost),
// then each thread emits out[v].
// ---------------------------------------------------------------------------
__global__ __launch_bounds__(1024)
void bp_fused(const float* __restrict__ l_v, const int* __restrict__ h,
              const float* __restrict__ s_c, const float* __restrict__ b,
              const float* __restrict__ w, const int* __restrict__ iters_p,
              const int* __restrict__ n1col, float* __restrict__ out,
              int E, int V) {
    const int v = threadIdx.x;

    __shared__ int srow[2];
    if (v < 2) srow[v] = 0;
    __syncthreads();
    atomicAdd(&srow[0], h[0 * V + v]);
    atomicAdd(&srow[1], h[1 * V + v]);
    __syncthreads();
    const int nr0 = srow[0];          // n1row[0] = sum_v h[0,v]
    const int nr1 = srow[1];          // n1row[1]

    const float E_f  = (float)E;
    const float nc0  = (float)n1col[0];
    const float nc1  = (float)n1col[1];
    const float w00  = w[0 * E + 0], w01 = w[0 * E + 1];
    const float w10  = w[1 * E + 0], w11 = w[1 * E + 1];
    const float base0 = l_v[0] * b[0];
    const float base1 = l_v[1] * b[1];
    const float sign0 = 1.0f - 2.0f * s_c[0];
    const float sign1 = 1.0f - 2.0f * s_c[1];
    const int   niter = iters_p[0];

    // mu_vc rows 0,1 take values {e=0, e=1, e>=2}; mu_cv rows 0,1 take {v=0, v=1, v>=2}
    float vc0_e0 = 0.f, vc0_e1 = 0.f, vc0_g = 0.f;
    float vc1_e0 = 0.f, vc1_e1 = 0.f, vc1_g = 0.f;
    float cv0_v0 = 0.f, cv0_v1 = 0.f, cv0_g = 0.f;
    float cv1_v0 = 0.f, cv1_v1 = 0.f, cv1_g = 0.f;

    for (int it = 0; it < niter; ++it) {
        // ---- v -> c (needs rows v=0,1), Jacobi: uses OLD mu_cv state
        float s00 = (E_f - nc0) * (cv0_v0 * w00);  // sub[0,0]
        float s01 = nc0 * (cv1_v0 * w01);          // sub[0,1]
        float tot0 = s00 + s01;
        float s10 = (E_f - nc1) * (cv0_v1 * w10);  // sub[1,0]
        float s11 = nc1 * (cv1_v1 * w11);          // sub[1,1]
        float tot1 = s10 + s11;
        float nvc0_e0 = tanhf(0.5f * (base0 + tot0 - s00));
        float nvc0_e1 = tanhf(0.5f * (base0 + tot0 - s01));
        float nvc0_g  = tanhf(0.5f * (base0 + tot0));
        float nvc1_e0 = tanhf(0.5f * (base1 + tot1 - s10));
        float nvc1_e1 = tanhf(0.5f * (base1 + tot1 - s11));
        float nvc1_g  = tanhf(0.5f * (base1 + tot1));

        // ---- c -> v (needs rows e=0,1), Jacobi: uses OLD mu_vc state
        float t0 = tanhf(0.5f * vc0_e0), t1 = tanhf(0.5f * vc1_e0);
        float p_v0 = ipow(t1, nr0);        // exclude h==0 terms -> t1^n1row
        float p_v1 = ipow(t0, V - nr0);    // exclude h==1 terms -> t0^n0row
        float p_g  = p_v0 * p_v1;          // no exclusion (v>=2)
        float ncv0_v0 = sign0 * 2.0f * atanhf(p_v0);
        float ncv0_v1 = sign0 * 2.0f * atanhf(p_v1);
        float ncv0_g  = sign0 * 2.0f * atanhf(p_g);

        float u0 = tanhf(0.5f * vc0_e1), u1 = tanhf(0.5f * vc1_e1);
        float q_v0 = ipow(u1, nr1);
        float q_v1 = ipow(u0, V - nr1);
        float q_g  = q_v0 * q_v1;
        float ncv1_v0 = sign1 * 2.0f * atanhf(q_v0);
        float ncv1_v1 = sign1 * 2.0f * atanhf(q_v1);
        float ncv1_g  = sign1 * 2.0f * atanhf(q_g);

        vc0_e0 = nvc0_e0; vc0_e1 = nvc0_e1; vc0_g = nvc0_g;
        vc1_e0 = nvc1_e0; vc1_e1 = nvc1_e1; vc1_g = nvc1_g;
        cv0_v0 = ncv0_v0; cv0_v1 = ncv0_v1; cv0_g = ncv0_g;
        cv1_v0 = ncv1_v0; cv1_v1 = ncv1_v1; cv1_g = ncv1_g;
    }

    // ---- marginalization (no exclusion)
    float cv0v = (v == 0) ? cv0_v0 : (v == 1) ? cv0_v1 : cv0_g;
    float cv1v = (v == 0) ? cv1_v0 : (v == 1) ? cv1_v1 : cv1_g;
    float ncv  = (float)n1col[v];
    float wv0  = w[v * E + 0], wv1 = w[v * E + 1];
    float tot  = (E_f - ncv) * (cv0v * wv0) + ncv * (cv1v * wv1);
    float mu   = tanhf(0.5f * (l_v[v] * b[v] + tot));
    out[v] = 1.0f / (expf(mu) + 1.0f);
}

extern "C" void kernel_launch(void* const* d_in, const int* in_sizes, int n_in,
                              void* d_out, int out_size, void* d_ws, size_t ws_size,
                              hipStream_t stream) {
    const float* l_v   = (const float*)d_in[0];
    const int*   h     = (const int*)  d_in[1];
    const float* s_c   = (const float*)d_in[2];
    const float* b     = (const float*)d_in[3];
    const float* w     = (const float*)d_in[4];
    const int*   iters = (const int*)  d_in[5];

    const int V = in_sizes[0];   // 512
    const int E = in_sizes[2];   // 256

    float* out   = (float*)d_out;
    int*   n1col = (int*)d_ws;   // V ints of scratch

    // 32 one-wave workgroups, 16 columns each, 4 accumulating WMMAs per wave.
    bp_count_cols_wmma<<<V / 16, 32, 0, stream>>>(h, n1col, E, V);
    // One block of V threads finishes everything.
    bp_fused<<<1, V, 0, stream>>>(l_v, h, s_c, b, w, iters, n1col, out, E, V);
}
